// DeformNet_43997644980911
// MI455X (gfx1250) — matrix operand order repaired and checked by reading, not compiled, not run on hardware
//
#include <hip/hip_runtime.h>
#include <hip/hip_fp16.h>

typedef __attribute__((ext_vector_type(16))) _Float16 v16h;
typedef __attribute__((ext_vector_type(8)))  float    v8f;

#define BB 8
#define NVV 2048
#define NPP 4096
#define DD 64
#define KK 8
#define TILE_M 16           // query vertices per workgroup (B columns)
#define TILE_K 128          // key rows streamed per iteration (8 waves x 16)
#define CROW  132           // padded LDS candidate row stride (u32)
#define IDXM  0xFFFu

union AOp { v16h v; _Float16 h[16]; uint4 q[2]; };
union COp { v8f v; float f[8]; };

__device__ __forceinline__ void cex(unsigned& a, unsigned& b) {
    const unsigned lo = a < b ? a : b;
    const unsigned hi = a < b ? b : a;
    a = lo; b = hi;
}

// Batcher odd-even mergesort, 8 elements, 19 comparators -> ascending
__device__ __forceinline__ void sort8(unsigned k[8]) {
    cex(k[0], k[1]); cex(k[2], k[3]); cex(k[0], k[2]); cex(k[1], k[3]); cex(k[1], k[2]);
    cex(k[4], k[5]); cex(k[6], k[7]); cex(k[4], k[6]); cex(k[5], k[7]); cex(k[5], k[6]);
    cex(k[0], k[4]); cex(k[2], k[6]); cex(k[2], k[4]);
    cex(k[1], k[5]); cex(k[3], k[7]); cex(k[3], k[5]);
    cex(k[1], k[2]); cex(k[3], k[4]); cex(k[5], k[6]);
}

// both lists ascending; keep top-8 of union in bv (ascending).
// max-pair of (asc, reversed-asc) is bitonic -> 3-stage bitonic sorter.
__device__ __forceinline__ void merge_top8(unsigned bv[8], const unsigned k[8]) {
    unsigned m[8];
    #pragma unroll
    for (int i = 0; i < 8; ++i) m[i] = bv[i] > k[7 - i] ? bv[i] : k[7 - i];
    cex(m[0], m[4]); cex(m[1], m[5]); cex(m[2], m[6]); cex(m[3], m[7]);
    cex(m[0], m[2]); cex(m[1], m[3]); cex(m[4], m[6]); cex(m[5], m[7]);
    cex(m[0], m[1]); cex(m[2], m[3]); cex(m[4], m[5]); cex(m[6], m[7]);
    #pragma unroll
    for (int i = 0; i < 8; ++i) bv[i] = m[i];
}

// ---------------- vm: sigmoid + per-batch min-max normalize ----------------
__global__ __launch_bounds__(256) void vm_kernel(const float* __restrict__ logits,
                                                 float* __restrict__ vm)
{
    __shared__ float s[NVV];
    __shared__ float rmin[256], rmax[256];
    const int b = blockIdx.x, tid = threadIdx.x;
    float lmin = 3.4e38f, lmax = -3.4e38f;
    for (int i = tid; i < NVV; i += 256) {
        float x  = logits[b * NVV + i];
        float sg = 1.0f / (1.0f + expf(-x));
        s[i] = sg;
        lmin = fminf(lmin, sg);
        lmax = fmaxf(lmax, sg);
    }
    rmin[tid] = lmin; rmax[tid] = lmax;
    __syncthreads();
    for (int off = 128; off > 0; off >>= 1) {
        if (tid < off) {
            rmin[tid] = fminf(rmin[tid], rmin[tid + off]);
            rmax[tid] = fmaxf(rmax[tid], rmax[tid + off]);
        }
        __syncthreads();
    }
    const float vmin = rmin[0];
    const float inv  = 1.0f / (rmax[0] - vmin);
    for (int i = tid; i < NVV; i += 256)
        vm[b * NVV + i] = (s[i] - vmin) * inv;
}

// ---------------- unit-normalize features -> f16 ----------------
__global__ __launch_bounds__(256) void unit_kernel(const float* __restrict__ vf,
                                                   const float* __restrict__ pf,
                                                   _Float16* __restrict__ vfn,
                                                   _Float16* __restrict__ pfn)
{
    const int r = blockIdx.x * blockDim.x + threadIdx.x;
    const int nv = BB * NVV, np = BB * NPP;
    if (r >= nv + np) return;
    const float* src;
    _Float16* dst;
    if (r < nv) { src = vf + (size_t)r * DD;        dst = vfn + (size_t)r * DD; }
    else        { src = pf + (size_t)(r - nv) * DD; dst = pfn + (size_t)(r - nv) * DD; }
    float ss = 0.f;
    #pragma unroll
    for (int j = 0; j < DD; ++j) { float x = src[j]; ss += x * x; }
    const float sc = 1.0f / fmaxf(sqrtf(ss), 1e-12f);
    #pragma unroll
    for (int j = 0; j < DD; ++j) dst[j] = (_Float16)(src[j] * sc);
}

// ------- fused WMMA similarity + in-register packed top-8 + weighted gather -------
// A (streamed) = key features, B (resident) = 16 query features.
// C layout: lane -> query column, vgpr -> key row => 8 candidates/lane in registers.
// MODE 0: queries=vtx, keys=pts  -> flow_init        (outbuf stride 3)
// MODE 1: queries=vtx, keys=vtx, keys masked by vis  (outbuf stride 4, final output)
template <int MODE>
__global__ __launch_bounds__(256) void knn_kernel(
    const _Float16* __restrict__ Qn,      // unit query feats f16 [B*NVV*64]
    const _Float16* __restrict__ Kn,      // unit key feats   f16 [B*NC *64]
    const float*    __restrict__ vfraw,   // raw query feats  f32
    const float*    __restrict__ kfraw,   // raw key feats    f32
    const float*    __restrict__ kpos,    // MODE0: pts [B*NC*3]; MODE1: flow_init
    const float*    __restrict__ vtx,     // MODE0: vtx positions
    const float*    __restrict__ vm,      // [B*NVV]
    const float*    __restrict__ flow_in, // MODE1: flow_init
    float*          __restrict__ outbuf,
    int NC)
{
    const int b    = blockIdx.y;
    const int row0 = blockIdx.x * TILE_M;    // query block
    const int tid  = threadIdx.x;
    const int wave = tid >> 5, lane = tid & 31;
    const int hl   = lane & 15, hi = lane >> 4;

    __shared__ unsigned s_cand[TILE_M * CROW];
    __shared__ float    s_acc[TILE_M * KK * 4];

    // --- B fragment: 16 query cols, resident across the whole loop ---
    const _Float16* qrow = Qn + (size_t)(b * NVV + row0 + hl) * DD;
    AOp qb0, qb1;
    qb0.q[0] = *(const uint4*)(qrow + hi * 8);
    qb0.q[1] = *(const uint4*)(qrow + hi * 8 + 16);
    qb1.q[0] = *(const uint4*)(qrow + hi * 8 + 32);
    qb1.q[1] = *(const uint4*)(qrow + hi * 8 + 48);

    // packed top-8, ascending (bv[0] = current min, bv[7] = best)
    unsigned bv[KK];
    #pragma unroll
    for (int j = 0; j < KK; ++j) bv[j] = 0u;

    const int nchunks = NC / TILE_K;         // 32 or 16 (even)
    const int rb = hi ? 8 : 0;

    // stream 16 key rows -> WMMA tile accumulator
    auto do_mm = [&](int ch, COp& c) {
        const int kr0 = ch * TILE_K + wave * 16;
        const _Float16* krow = Kn + (size_t)(b * NC + kr0 + hl) * DD;
        const int e0 = hi ? 8 : 0;
        AOp ka0, ka1;
        ka0.q[0] = *(const uint4*)(krow + e0);
        ka0.q[1] = *(const uint4*)(krow + e0 + 16);
        ka1.q[0] = *(const uint4*)(krow + e0 + 32);
        ka1.q[1] = *(const uint4*)(krow + e0 + 48);
        #pragma unroll
        for (int i = 0; i < 8; ++i) c.f[i] = 0.f;
        c.v = __builtin_amdgcn_wmma_f32_16x16x32_f16(false, ka0.v, false, qb0.v,
                                                     (short)0, c.v, false, false);
        c.v = __builtin_amdgcn_wmma_f32_16x16x32_f16(false, ka1.v, false, qb1.v,
                                                     (short)0, c.v, false, false);
    };

    // pack 8 candidates, sort, merge into running top-8
    auto do_sel = [&](int ch, COp& c) {
        const int kr0 = ch * TILE_K + wave * 16;
        float vmv[8];
        if (MODE == 1) {
            const float4 va  = *(const float4*)(vm + b * NVV + kr0 + rb);
            const float4 vb4 = *(const float4*)(vm + b * NVV + kr0 + rb + 4);
            vmv[0] = va.x;  vmv[1] = va.y;  vmv[2] = va.z;  vmv[3] = va.w;
            vmv[4] = vb4.x; vmv[5] = vb4.y; vmv[6] = vb4.z; vmv[7] = vb4.w;
        }
        unsigned k[8];
        #pragma unroll
        for (int i = 0; i < 8; ++i) {
            const unsigned kidx = (unsigned)(kr0 + rb + i);
            unsigned key = (__float_as_uint(c.f[i] + 2.0f) & ~IDXM) | (kidx ^ IDXM);
            if (MODE == 1) {
                if (vmv[i] < 0.5f) key = kidx ^ IDXM;  // value bits 0: below all visible
            }
            k[i] = key;
        }
        sort8(k);
        merge_top8(bv, k);
    };

    // --- software-pipelined main loop: WMMA of chunk ch overlaps selection of ch-1 ---
    COp cA, cB;
    do_mm(0, cA);
    int ch = 1;
    for (; ch + 1 < nchunks; ch += 2) {
        do_mm(ch, cB);
        do_sel(ch - 1, cA);
        do_mm(ch + 1, cA);
        do_sel(ch, cB);
    }
    do_mm(nchunks - 1, cB);
    do_sel(nchunks - 2, cA);
    do_sel(nchunks - 1, cB);

    // --- dump per-lane sorted lists: 16 lists of 8 per query column ---
    const int l = wave * 2 + hi;             // list id 0..15
    #pragma unroll
    for (int j = 0; j < KK; ++j) s_cand[hl * CROW + l * KK + j] = bv[j];
    __syncthreads();

    // --- parallel tree merge: 16 sorted lists -> top-8 per query ---
    #pragma unroll
    for (int step = 1; step < 16; step <<= 1) {
        if ((l & (2 * step - 1)) == 0) {
            unsigned p[8];
            #pragma unroll
            for (int j = 0; j < KK; ++j) p[j] = s_cand[hl * CROW + (l + step) * KK + j];
            merge_top8(bv, p);
            #pragma unroll
            for (int j = 0; j < KK; ++j) s_cand[hl * CROW + l * KK + j] = bv[j];
        }
        __syncthreads();
    }
    // s_cand[q*CROW + 0..7] now holds the top-8 packed keys (index embedded)

    // --- gather K neighbors, fsim-weighted accumulate (raw fp32 features) ---
    if (tid < TILE_M * KK) {
        const int row = tid >> 3, k = tid & 7;
        const int rg  = b * NVV + row0 + row;
        const int n   = (int)((s_cand[row * CROW + k] & IDXM) ^ IDXM);
        const float* kfeat = kfraw + (size_t)(b * NC + n) * DD;
        const float* vfeat = vfraw + (size_t)rg * DD;
        float dot = 0.f;
        #pragma unroll
        for (int j = 0; j < DD; ++j) dot += kfeat[j] * vfeat[j];
        float w, px, py, pz;
        if (MODE == 0) {
            w = dot * vm[rg];
            const float* p  = kpos + (size_t)(b * NC + n) * 3;
            const float* qv = vtx  + (size_t)rg * 3;
            px = p[0] - qv[0]; py = p[1] - qv[1]; pz = p[2] - qv[2];
        } else {
            w = dot;
            const float* p = kpos + (size_t)(b * NVV + n) * 3;  // neighbor flow_init
            px = p[0]; py = p[1]; pz = p[2];
        }
        s_acc[tid * 4 + 0] = px * w;
        s_acc[tid * 4 + 1] = py * w;
        s_acc[tid * 4 + 2] = pz * w;
        s_acc[tid * 4 + 3] = w;
    }
    __syncthreads();
    if (tid < TILE_M) {
        const int row = tid;
        const int rg  = b * NVV + row0 + row;
        float nx = 0, ny = 0, nz = 0, dw = 0;
        #pragma unroll
        for (int k = 0; k < KK; ++k) {
            nx += s_acc[(row * KK + k) * 4 + 0];
            ny += s_acc[(row * KK + k) * 4 + 1];
            nz += s_acc[(row * KK + k) * 4 + 2];
            dw += s_acc[(row * KK + k) * 4 + 3];
        }
        const float inv = 1.0f / dw;
        if (MODE == 0) {
            outbuf[(size_t)rg * 3 + 0] = nx * inv;
            outbuf[(size_t)rg * 3 + 1] = ny * inv;
            outbuf[(size_t)rg * 3 + 2] = nz * inv;
        } else {
            const float v   = vm[rg];
            const bool  vis = v >= 0.5f;
            outbuf[(size_t)rg * 4 + 0] = vis ? flow_in[(size_t)rg * 3 + 0] : nx * inv;
            outbuf[(size_t)rg * 4 + 1] = vis ? flow_in[(size_t)rg * 3 + 1] : ny * inv;
            outbuf[(size_t)rg * 4 + 2] = vis ? flow_in[(size_t)rg * 3 + 2] : nz * inv;
            outbuf[(size_t)rg * 4 + 3] = v;
        }
    }
}

extern "C" void kernel_launch(void* const* d_in, const int* in_sizes, int n_in,
                              void* d_out, int out_size, void* d_ws, size_t ws_size,
                              hipStream_t stream) {
    const float* vtx = (const float*)d_in[0];   // [B*NV,3]
    const float* pts = (const float*)d_in[1];   // [B*NP,3]
    const float* vf  = (const float*)d_in[2];   // [B*NV,64]
    const float* pf  = (const float*)d_in[3];   // [B*NP,64]
    const float* vml = (const float*)d_in[4];   // [B*NV,1]
    float* out = (float*)d_out;                 // [B*NV,4]

    // workspace layout
    float*    vm   = (float*)d_ws;                               // B*NV
    float*    flow = vm + (size_t)BB * NVV;                      // B*NV*3
    _Float16* vfn  = (_Float16*)(flow + (size_t)BB * NVV * 3);   // B*NV*64 f16
    _Float16* pfn  = vfn + (size_t)BB * NVV * DD;                // B*NP*64 f16

    vm_kernel<<<dim3(BB), dim3(256), 0, stream>>>(vml, vm);

    const int totrows = BB * NVV + BB * NPP;
    unit_kernel<<<dim3((totrows + 255) / 256), dim3(256), 0, stream>>>(vf, pf, vfn, pfn);

    dim3 grid(NVV / TILE_M, BB);
    knn_kernel<0><<<grid, dim3(256), 0, stream>>>(vfn, pfn, vf, pf, pts, vtx, vm, flow,
                                                  flow, NPP);
    knn_kernel<1><<<grid, dim3(256), 0, stream>>>(vfn, vfn, vf, vf, flow, vtx, vm, flow,
                                                  out, NVV);
}
